// DeepHamCritic_58222576664665
// MI455X (gfx1250) — matrix operand order repaired and checked by reading, not compiled, not run on hardware
//
#include <hip/hip_runtime.h>
#include <hip/hip_bf16.h>

typedef __attribute__((ext_vector_type(16))) __bf16 v16bf;
typedef __attribute__((ext_vector_type(8)))  __bf16 v8bf;
typedef __attribute__((ext_vector_type(8)))  float  v8f;

#define ALPHA 0.1f

// pack two f32 -> one dword of 2 bf16 (v_cvt_pk_bf16_f32)
static __device__ inline unsigned pack2bf(float a, float b) {
    __hip_bfloat162 p = __float22bfloat162_rn(make_float2(a, b));
    union { __hip_bfloat162 h; unsigned u; } cv;
    cv.h = p;
    return cv.u;
}

// build a v16bf from two aligned 16-byte LDS chunks (2x ds_load_b128)
static __device__ inline v16bf load_frag_2x8(const __bf16* lo_p, const __bf16* hi_p) {
    v8bf lo = *reinterpret_cast<const v8bf*>(lo_p);
    v8bf hi = *reinterpret_cast<const v8bf*>(hi_p);
    v16bf r;
    #pragma unroll
    for (int j = 0; j < 8; ++j) { r[j] = lo[j]; r[j + 8] = hi[j]; }
    return r;
}

// ---------------------------------------------------------------------------
// Tiled GEMM: C[M,Nc] = act(A[M,K] @ W[K,Nc] + bias), f32 in/out, bf16 WMMA.
// Block = 256 threads (8 wave32). Tile = 128x128, K-step = 32.
// Waves 4(M) x 2(N); each wave computes 32x64 as 2x4 WMMA 16x16x32 -> 8 wmma
// per wave per K-step against 12 b128 LDS fragment loads.
// B is staged *swizzled*: element (k,n) lands at [cb=n/16][L=(n%16)+16*(k>=16)]
// [j=k%16] so a consuming lane reads 16 contiguous bf16 (two aligned 16B chunks).
// ---------------------------------------------------------------------------
__global__ __launch_bounds__(256)
void gemm_bf16_kernel(const float* __restrict__ A, const float* __restrict__ W,
                      const float* __restrict__ bias, float* __restrict__ C,
                      int M, int K, int Nc, int doLeaky) {
    // A: [row][k] packed pairs, row stride 20 dwords (80B) -> 16B-aligned frag chunks
    __shared__ __attribute__((aligned(16))) unsigned sAu[128 * 20];  // 10 KB
    // B: swizzled fragment-major, lane stride 24 bf16 (48B), cb stride 768 bf16
    __shared__ __attribute__((aligned(16))) __bf16  sB[8 * 768];     // 12 KB

    const int t    = threadIdx.x;
    const int lane = t & 31;
    const int wid  = t >> 5;
    const int wm   = (wid & 3) * 32;    // wave row offset
    const int wn   = (wid >> 2) * 64;   // wave col offset
    const int cb0  = (wid >> 2) * 4;    // wave col-block base (16-col blocks)

    const int n0 = blockIdx.x * 128;
    const int m0 = blockIdx.y * 128;

    v8f acc[2][4] = {};

    const int lr     = lane & 15;
    const int kbaseA = (lane >> 4) * 16;  // A fragment K base (contiguous k = 16h + s, matches swizzled B)

    for (int k0 = 0; k0 < K; k0 += 32) {
        // ---- stage A tile: 128x32 f32 -> packed bf16 pairs (4 float4/thread)
        #pragma unroll
        for (int i = 0; i < 4; ++i) {
            int idx = t + i * 256;            // 0..1023 float4s
            int row = idx >> 3;               // 0..127
            int c4  = idx & 7;                // 0..7
            int gr  = m0 + row;
            float4 v = make_float4(0.f, 0.f, 0.f, 0.f);
            if (gr < M)
                v = *reinterpret_cast<const float4*>(A + (size_t)gr * K + k0 + c4 * 4);
            sAu[row * 20 + c4 * 2 + 0] = pack2bf(v.x, v.y);
            sAu[row * 20 + c4 * 2 + 1] = pack2bf(v.z, v.w);
        }
        // ---- stage B tile: 32x128 f32 -> swizzled bf16 (8 k-pairs/thread)
        #pragma unroll
        for (int i = 0; i < 8; ++i) {
            int p  = t + i * 256;             // 0..2047 k-pairs
            int n  = p & 127;
            int kp = p >> 7;                  // 0..15, k = 2*kp
            const float* wp = W + (size_t)(k0 + 2 * kp) * Nc + n0 + n;
            float w0 = wp[0];
            float w1 = wp[Nc];
            int cb = n >> 4;
            int L  = (n & 15) + ((kp >= 8) ? 16 : 0);
            int j  = (2 * kp) & 15;
            reinterpret_cast<unsigned*>(sB)[cb * 384 + L * 12 + (j >> 1)] = pack2bf(w0, w1);
        }
        // ---- prefetch next K-tile into L2 while WMMAs run
        if (k0 + 32 < K) {
            if (t < 128) {
                int gr = m0 + t;
                if (gr < M) __builtin_prefetch(A + (size_t)gr * K + k0 + 32, 0, 3);
            } else {
                int r   = (t - 128) >> 2;
                int seg = (t - 128) & 3;
                __builtin_prefetch(W + (size_t)(k0 + 32 + r) * Nc + n0 + seg * 32, 0, 3);
            }
        }
        __syncthreads();

        // ---- fragments (all reads are aligned 16B ds_load_b128 pairs)
        v16bf afrag[2], bfrag[4];
        #pragma unroll
        for (int tm = 0; tm < 2; ++tm) {
            const __bf16* ap =
                reinterpret_cast<const __bf16*>(sAu + (wm + tm * 16 + lr) * 20);
            afrag[tm] = load_frag_2x8(ap + kbaseA, ap + kbaseA + 8);
        }
        #pragma unroll
        for (int tn = 0; tn < 4; ++tn) {
            const __bf16* bp = sB + (cb0 + tn) * 768 + lane * 24;
            bfrag[tn] = load_frag_2x8(bp, bp + 8);
        }

        #pragma unroll
        for (int tm = 0; tm < 2; ++tm)
            #pragma unroll
            for (int tn = 0; tn < 4; ++tn)
                acc[tm][tn] = __builtin_amdgcn_wmma_f32_16x16x32_bf16(
                    false, afrag[tm], false, bfrag[tn],
                    (short)0, acc[tm][tn], false, false);

        __syncthreads();
    }

    // ---- epilogue: bias + leaky (C/D layout: M = r + 8*(lane>=16), N = lane&15)
    const int rowoff = (lane >> 4) * 8;
    #pragma unroll
    for (int tm = 0; tm < 2; ++tm) {
        #pragma unroll
        for (int tn = 0; tn < 4; ++tn) {
            int col = n0 + wn + tn * 16 + lr;
            float bv = bias ? bias[col] : 0.f;
            #pragma unroll
            for (int r = 0; r < 8; ++r) {
                int row = m0 + wm + tm * 16 + rowoff + r;
                if (row < M) {
                    float v = acc[tm][tn][r] + bv;
                    if (doLeaky) v = (v > 0.f) ? v : ALPHA * v;
                    C[(size_t)row * Nc + col] = v;
                }
            }
        }
    }
}

// ---------------------------------------------------------------------------
// Graph aggregation helpers
// ---------------------------------------------------------------------------
__global__ void fill_kernel(float* __restrict__ p, float v, int n) {
    int stride = gridDim.x * blockDim.x;
    for (int i = blockIdx.x * blockDim.x + threadIdx.x; i < n; i += stride)
        p[i] = v;
}

__global__ void deg_kernel(const long long* __restrict__ dst, float* __restrict__ deg, int E) {
    int stride = gridDim.x * blockDim.x;
    for (int e = blockIdx.x * blockDim.x + threadIdx.x; e < E; e += stride)
        atomicAdd(&deg[(int)dst[e]], 1.0f);
}

__global__ void rsqrt_kernel(float* __restrict__ dinv, int n) {
    int stride = gridDim.x * blockDim.x;
    for (int i = blockIdx.x * blockDim.x + threadIdx.x; i < n; i += stride)
        dinv[i] = rsqrtf(dinv[i]);   // deg >= 1 always (self loop)
}

// agg[n,:] = h[n,:] * dinv[n]^2 + b_gcn  (self-loop contribution + bias)
__global__ void init_agg_kernel(const float* __restrict__ h, const float* __restrict__ dinv,
                                const float* __restrict__ bgcn, float* __restrict__ agg, int Nn) {
    int total  = Nn * 128;                 // float4 chunks of 512-wide rows
    int stride = gridDim.x * blockDim.x;
    for (int idx = blockIdx.x * blockDim.x + threadIdx.x; idx < total; idx += stride) {
        int node = idx >> 7;
        int c4   = idx & 127;
        float di = dinv[node];
        float w  = di * di;
        float4 v = reinterpret_cast<const float4*>(h + (size_t)node * 512)[c4];
        float4 b = reinterpret_cast<const float4*>(bgcn)[c4];
        float4 o;
        o.x = v.x * w + b.x; o.y = v.y * w + b.y;
        o.z = v.z * w + b.z; o.w = v.w * w + b.w;
        reinterpret_cast<float4*>(agg + (size_t)node * 512)[c4] = o;
    }
}

// One wave per edge: coalesced float4 gather of h[src], f32 atomic scatter to
// agg[dst]. agg (102 MB) fits in the 192 MB L2, so atomics land in-cache.
__global__ void edge_msg_kernel(const long long* __restrict__ src, const long long* __restrict__ dst,
                                const float* __restrict__ h, const float* __restrict__ dinv,
                                float* __restrict__ agg, int E) {
    int gw    = (blockIdx.x * blockDim.x + threadIdx.x) >> 5;
    int lane  = threadIdx.x & 31;
    int nwave = (gridDim.x * blockDim.x) >> 5;
    for (int e = gw; e < E; e += nwave) {
        int s = (int)src[e];
        int d = (int)dst[e];
        float w = dinv[s] * dinv[d];
        const float4* hp = reinterpret_cast<const float4*>(h + (size_t)s * 512);
        float* ap = agg + (size_t)d * 512;
        #pragma unroll
        for (int i = 0; i < 4; ++i) {
            int c4 = lane + i * 32;        // 0..127
            float4 v = hp[c4];
            atomicAdd(ap + c4 * 4 + 0, v.x * w);
            atomicAdd(ap + c4 * 4 + 1, v.y * w);
            atomicAdd(ap + c4 * 4 + 2, v.z * w);
            atomicAdd(ap + c4 * 4 + 3, v.w * w);
        }
    }
}

// out[n] = dot(h3[n,:256], W_out) + b_out  -- one wave per node
__global__ void out_kernel(const float* __restrict__ h3, const float* __restrict__ Wout,
                           const float* __restrict__ bout, float* __restrict__ out, int M) {
    int node = (blockIdx.x * blockDim.x + threadIdx.x) >> 5;
    int lane = threadIdx.x & 31;
    if (node >= M) return;
    const float* row = h3 + (size_t)node * 256;
    float s = 0.f;
    #pragma unroll
    for (int j = 0; j < 8; ++j) {
        int c = lane + j * 32;
        s += row[c] * Wout[c];
    }
    #pragma unroll
    for (int off = 16; off > 0; off >>= 1)
        s += __shfl_xor(s, off, 32);
    if (lane == 0) out[node] = s + bout[0];
}

// ---------------------------------------------------------------------------
// Orchestration
// ---------------------------------------------------------------------------
extern "C" void kernel_launch(void* const* d_in, const int* in_sizes, int n_in,
                              void* d_out, int out_size, void* d_ws, size_t ws_size,
                              hipStream_t stream) {
    const float*     x     = (const float*)d_in[0];
    const long long* edge  = (const long long*)d_in[1];   // [2,E] int64
    const float*     Wgcn  = (const float*)d_in[2];
    const float*     bgcn  = (const float*)d_in[3];
    const float*     W1    = (const float*)d_in[4];
    const float*     b1    = (const float*)d_in[5];
    const float*     W2    = (const float*)d_in[6];
    const float*     b2    = (const float*)d_in[7];
    const float*     W3    = (const float*)d_in[8];
    const float*     b3    = (const float*)d_in[9];
    const float*     Wout  = (const float*)d_in[10];
    const float*     bout  = (const float*)d_in[11];
    float*           out   = (float*)d_out;

    const int N = in_sizes[0] / 128;      // 50000 nodes
    const int E = in_sizes[1] / 2;        // 800000 edges
    const int D = 128, C0 = 512, H = 256;

    const long long* srcIdx = edge;
    const long long* dstIdx = edge + E;

    // Workspace layout (ping-pong): bufA [N,512], bufB [N,512], dinv [N]
    float* bufA = (float*)d_ws;                       // h, then h1 (front), h3 (back half)
    float* bufB = bufA + (size_t)N * C0;              // agg, then h2
    float* dinv = bufB + (size_t)N * C0;

    float* h   = bufA;
    float* agg = bufB;
    float* h1  = bufA;                                // [N,256] overwrites h (done)
    float* h2  = bufB;                                // [N,256] overwrites agg (done)
    float* h3  = bufA + (size_t)N * H;                // [N,256] in back half of bufA

    // 1) degrees: dinv = 1 (self loop) + in-degree, then rsqrt
    fill_kernel<<<1024, 256, 0, stream>>>(dinv, 1.0f, N);
    deg_kernel<<<2048, 256, 0, stream>>>(dstIdx, dinv, E);
    rsqrt_kernel<<<1024, 256, 0, stream>>>(dinv, N);

    // 2) h = x @ W_gcn   (bias deferred to aggregation init)
    {
        dim3 grid(C0 / 128, (N + 127) / 128);
        gemm_bf16_kernel<<<grid, 256, 0, stream>>>(x, Wgcn, nullptr, h, N, D, C0, 0);
    }

    // 3) agg = h*dinv^2 + b_gcn, then scatter edge messages (atomics hit L2)
    init_agg_kernel<<<4096, 256, 0, stream>>>(h, dinv, bgcn, agg, N);
    edge_msg_kernel<<<2048, 256, 0, stream>>>(srcIdx, dstIdx, h, dinv, agg, E);

    // 4) MLP: three fused GEMM+bias+leaky layers on WMMA
    {
        dim3 grid(H / 128, (N + 127) / 128);
        gemm_bf16_kernel<<<grid, 256, 0, stream>>>(agg, W1, b1, h1, N, C0, H, 1);
        gemm_bf16_kernel<<<grid, 256, 0, stream>>>(h1,  W2, b2, h2, N, H,  H, 1);
        gemm_bf16_kernel<<<grid, 256, 0, stream>>>(h2,  W3, b3, h3, N, H,  H, 1);
    }

    // 5) final projection 256 -> 1
    out_kernel<<<(N * 32 + 255) / 256, 256, 0, stream>>>(h3, Wout, bout, out, N);

    (void)n_in; (void)out_size; (void)ws_size;
}
